// FarmacoNetCompleta_48790828482639
// MI455X (gfx1250) — matrix operand order repaired and checked by reading, not compiled, not run on hardware
//
#include <hip/hip_runtime.h>
#include <hip/hip_bf16.h>

typedef __attribute__((ext_vector_type(16))) _Float16 v16h;
typedef __attribute__((ext_vector_type(2)))  _Float16 h2;
typedef __attribute__((ext_vector_type(8)))  float    v8f;

#define HDIM 64
#define NGRAPH 1024

// ---------------- helpers ----------------
__global__ void zero_f32(float* __restrict__ p, int n) {
  int t = blockIdx.x * blockDim.x + threadIdx.x;
  if (t < n) p[t] = 0.0f;
}

__global__ void deg_count(const int* __restrict__ dst, float* __restrict__ deg, int E) {
  int t = blockIdx.x * blockDim.x + threadIdx.x;
  if (t < E) atomicAdd(&deg[dst[t]], 1.0f);
}

__global__ void deg_to_dinv(float* __restrict__ deg, int N) {
  int t = blockIdx.x * blockDim.x + threadIdx.x;
  if (t < N) deg[t] = rsqrtf(deg[t] + 1.0f);   // +1 self loop
}

// ---------------- WMMA GEMM: out[N,64] = (maybe-ReLU h)[N,K] @ W[K,64] ----------------
// Block = 256 threads = 8 waves; W staged once per block into LDS (f16, transposed+padded).
// One wave -> 16 rows x 64 cols, 4 accumulators of 16x16, K stepped by 32.
template <bool RELU_IN>
__global__ void gemm_wmma(const float* __restrict__ h, const float* __restrict__ W,
                          float* __restrict__ out, int N, int K) {
  // Transposed weight tile: Wt[n][k], padded stride (K+2 halves) to kill bank conflicts.
  __shared__ _Float16 Wt[HDIM * (128 + 2)];
  const int KP  = K + 2;
  const int tid = threadIdx.x;
  for (int idx = tid; idx < K * HDIM; idx += blockDim.x) {
    const int k = idx >> 6;        // idx / HDIM
    const int n = idx & 63;        // idx % HDIM
    Wt[n * KP + k] = (_Float16)W[idx];
  }
  __syncthreads();

  const int lane = tid & 31;
  const int wave = tid >> 5;
  const int row0 = (blockIdx.x * (blockDim.x >> 5) + wave) * 16;
  if (row0 >= N) return;                        // wave-uniform, after the barrier

  const int hs = lane >> 4;                     // half-wave select (0/1)
  const int nl = lane & 15;
  // A-matrix: M = nl for both lane halves; clamp gather row for the partial tile
  const long long rowm = (row0 + nl < N) ? (row0 + nl) : (N - 1);
  const float* __restrict__ hrow = h + rowm * (long long)K;

  v8f acc[4] = {};                              // n0 = 0,16,32,48

  for (int kb = 0; kb < K; kb += 32) {
    // ---- A fragment 16x32 f16 (contiguous k = 16*hs + s, matching B): float2 global loads ----
    v16h a;
#pragma unroll
    for (int v = 0; v < 8; ++v) {
      const int k = kb + hs * 16 + 2 * v;
      float2 f = *(const float2*)(hrow + k);
      if (RELU_IN) { f.x = fmaxf(f.x, 0.0f); f.y = fmaxf(f.y, 0.0f); }
      a[2 * v]     = (_Float16)f.x;
      a[2 * v + 1] = (_Float16)f.y;
    }
    // ---- B fragments 32x16 f16 from LDS: one ds_load_b32 per K-pair ----
#pragma unroll
    for (int t = 0; t < 4; ++t) {
      v16h b;
      const _Float16* __restrict__ wcol = Wt + (t * 16 + nl) * KP + kb + hs * 16;
#pragma unroll
      for (int v = 0; v < 8; ++v) {
        const h2 p = *(const h2*)(wcol + 2 * v);   // 4-byte aligned pair (k, k+1)
        b[2 * v]     = p.x;
        b[2 * v + 1] = p.y;
      }
      acc[t] = __builtin_amdgcn_wmma_f32_16x16x32_f16(
          /*neg_a=*/false, a, /*neg_b=*/false, b,
          /*c_mod=*/(short)0, acc[t], /*reuse_a=*/false, /*reuse_b=*/false);
    }
  }

  // ---- store: C/D layout — lane gives N column, VGPR r gives M (lanes 16-31: M=r+8) ----
  float* __restrict__ obase = out + (long long)row0 * HDIM + nl;
  if (row0 + 16 <= N) {                         // full tile: unpredicated stores
#pragma unroll
    for (int t = 0; t < 4; ++t)
#pragma unroll
      for (int r = 0; r < 8; ++r)
        obase[(long long)(hs * 8 + r) * HDIM + t * 16] = acc[t][r];
  } else {                                      // tail tile only
#pragma unroll
    for (int t = 0; t < 4; ++t)
#pragma unroll
      for (int r = 0; r < 8; ++r) {
        const int row = row0 + hs * 8 + r;
        if (row < N) obase[(long long)(hs * 8 + r) * HDIM + t * 16] = acc[t][r];
      }
  }
}

// out[i,:] = hw[i,:] * dinv[i]^2 + bias   (self-loop message + bias; also initializes buffer)
__global__ void init_out(const float* __restrict__ hw, const float* __restrict__ dinv,
                         const float* __restrict__ bias, float* __restrict__ out, int N) {
  int t = blockIdx.x * blockDim.x + threadIdx.x;
  int i = t >> 6, c = t & 63;
  if (i < N) {
    float di = dinv[i];
    out[t] = hw[t] * di * di + bias[c];
  }
}

// out[dst] += hw[src] * dinv[src]*dinv[dst]; 16 threads per edge, float4 per thread
__global__ void edge_msg(const int* __restrict__ src, const int* __restrict__ dst,
                         const float* __restrict__ dinv, const float* __restrict__ hw,
                         float* __restrict__ out, int E) {
  long long t = (long long)blockIdx.x * blockDim.x + threadIdx.x;
  long long e = t >> 4;
  if (e >= E) return;
  const int c0 = (int)(t & 15) * 4;
  const int s = src[e], d = dst[e];
  const float w = dinv[s] * dinv[d];
  const float4 v = *(const float4*)(hw + (long long)s * HDIM + c0);
  float* o = out + (long long)d * HDIM + c0;
  atomicAdd(o + 0, v.x * w);
  atomicAdd(o + 1, v.y * w);
  atomicAdd(o + 2, v.z * w);
  atomicAdd(o + 3, v.w * w);
}

__global__ void pool_scatter(const float* __restrict__ h, const int* __restrict__ batch,
                             float* __restrict__ sums, float* __restrict__ cnts, int N) {
  int t = blockIdx.x * blockDim.x + threadIdx.x;
  int i = t >> 6, c = t & 63;
  if (i < N) {
    int g = batch[i];
    atomicAdd(&sums[g * HDIM + c], h[t]);
    if (c == 0) atomicAdd(&cnts[g], 1.0f);
  }
}

__global__ void final_linear(const float* __restrict__ sums, const float* __restrict__ cnts,
                             const float* __restrict__ lw, const float* __restrict__ lb,
                             float* __restrict__ out) {
  int g = blockIdx.x * blockDim.x + threadIdx.x;
  if (g < NGRAPH) {
    float inv = 1.0f / fmaxf(cnts[g], 1.0f);
    float acc = 0.0f;
#pragma unroll
    for (int c = 0; c < HDIM; ++c) acc += sums[g * HDIM + c] * inv * lw[c];
    out[g] = acc + lb[0];
  }
}

// ---------------- launch ----------------
extern "C" void kernel_launch(void* const* d_in, const int* in_sizes, int n_in,
                              void* d_out, int out_size, void* d_ws, size_t ws_size,
                              hipStream_t stream) {
  const float* x    = (const float*)d_in[0];
  const int*   ei   = (const int*)  d_in[1];   // [2, E]
  const int*   batch= (const int*)  d_in[3];
  const float* W1 = (const float*)d_in[4];  const float* b1 = (const float*)d_in[5];
  const float* W2 = (const float*)d_in[6];  const float* b2 = (const float*)d_in[7];
  const float* W3 = (const float*)d_in[8];  const float* b3 = (const float*)d_in[9];
  const float* lw = (const float*)d_in[10]; const float* lb = (const float*)d_in[11];
  float* out = (float*)d_out;

  const int N = in_sizes[0] / 128;   // F_IN = 128
  const int E = in_sizes[1] / 2;
  const int* src = ei;
  const int* dst = ei + E;

  // workspace layout (floats): dinv[N] | bufA[N*64] | bufB[N*64] | sums[1024*64] | cnts[1024]
  float* ws   = (float*)d_ws;
  float* dinv = ws;
  float* bufA = ws + N;
  float* bufB = bufA + (size_t)N * HDIM;
  float* sums = bufB + (size_t)N * HDIM;
  float* cnts = sums + (size_t)NGRAPH * HDIM;

  const int B = 256;
  const int nBlk   = (N + B - 1) / B;
  const int eBlk   = (E + B - 1) / B;
  const int nhBlk  = (int)(((size_t)N * HDIM + B - 1) / B);
  const int emBlk  = (int)(((size_t)E * 16 + B - 1) / B);
  const int gemmBlk = ((N + 15) / 16 + 7) / 8;   // 8 waves/block, 16 rows/wave

  // degree / normalization
  zero_f32   <<<nBlk, B, 0, stream>>>(dinv, N);
  deg_count  <<<eBlk, B, 0, stream>>>(dst, dinv, E);
  deg_to_dinv<<<nBlk, B, 0, stream>>>(dinv, N);

  // layer 1 (input x, K=128)
  gemm_wmma<false><<<gemmBlk, 256, 0, stream>>>(x, W1, bufA, N, 128);
  init_out <<<nhBlk, B, 0, stream>>>(bufA, dinv, b1, bufB, N);
  edge_msg <<<emBlk, B, 0, stream>>>(src, dst, dinv, bufA, bufB, E);

  // layer 2 (ReLU fused into A-fragment gather, K=64)
  gemm_wmma<true><<<gemmBlk, 256, 0, stream>>>(bufB, W2, bufA, N, HDIM);
  init_out<<<nhBlk, B, 0, stream>>>(bufA, dinv, b2, bufB, N);
  edge_msg<<<emBlk, B, 0, stream>>>(src, dst, dinv, bufA, bufB, E);

  // layer 3 (ReLU on input; no ReLU on output)
  gemm_wmma<true><<<gemmBlk, 256, 0, stream>>>(bufB, W3, bufA, N, HDIM);
  init_out<<<nhBlk, B, 0, stream>>>(bufA, dinv, b3, bufB, N);
  edge_msg<<<emBlk, B, 0, stream>>>(src, dst, dinv, bufA, bufB, E);

  // global mean pool + final linear
  zero_f32<<<(NGRAPH * HDIM + NGRAPH + B - 1) / B, B, 0, stream>>>(sums, NGRAPH * HDIM + NGRAPH);
  pool_scatter<<<nhBlk, B, 0, stream>>>(bufB, batch, sums, cnts, N);
  final_linear<<<(NGRAPH + B - 1) / B, B, 0, stream>>>(sums, cnts, lw, lb, out);
}
